// TopiCOT_66606352826421
// MI455X (gfx1250) — compile-verified, matrix operand order checked by reading, not run on hardware
//
#include <hip/hip_runtime.h>
#include <hip/hip_bf16.h>
#include <math.h>

// ---------------------------------------------------------------------------
// Problem constants (match reference)
// ---------------------------------------------------------------------------
#define B_   1024
#define V_   50000
#define K_   100
#define EU_  200
#define E_   200
#define G_   20
#define P_   384
#define SK_EPS   1e-16f
#define BN_EPS   1e-5f
#define ALPHA_   20.0f

typedef __attribute__((ext_vector_type(16))) __bf16 v16bf;
typedef __attribute__((ext_vector_type(8)))  float  v8f;

// ---------------------------------------------------------------------------
// WMMA fragment-major LDS helpers.
// ISA 7.12.2 (16-bit A/B 16x32): lane = h*16 + idx16, per-lane element e:
//   e<8:  K = h*8 + e ;  e>=8: K = 16 + h*8 + (e-8)
// Inverse: K -> (h, e). Even/odd K pairs share a slot at (e, e+1), so staging
// packs two bf16 per 32-bit LDS store, and fragment loads are one v16bf
// (32B contiguous) -> 2x ds_load_b128 instead of 16x ds_load_u16.
// ---------------------------------------------------------------------------
__device__ __forceinline__ void fragpos(int k, int& h, int& e) {
  int hi = (k >> 4) & 1;
  int kk = k & 15;
  h = kk >> 3;
  e = (hi ? 8 : 0) + (kk & 7);
}

__device__ __forceinline__ unsigned pack_bf16(float a, float b) {
  __bf16 x = (__bf16)a, y = (__bf16)b;
  unsigned short ux = __builtin_bit_cast(unsigned short, x);
  unsigned short uy = __builtin_bit_cast(unsigned short, y);
  return ((unsigned)uy << 16) | (unsigned)ux;
}

__device__ __forceinline__ float softplusf(float x) {
  return x > 20.f ? x : log1pf(__expf(x));
}

// Block (256-thread) reductions, result valid in thread 0.
__device__ __forceinline__ float blockReduceSum256(float v) {
  __shared__ float tmp[8];
  __syncthreads();
  v += __shfl_xor(v, 16); v += __shfl_xor(v, 8); v += __shfl_xor(v, 4);
  v += __shfl_xor(v, 2);  v += __shfl_xor(v, 1);
  if ((threadIdx.x & 31) == 0) tmp[threadIdx.x >> 5] = v;
  __syncthreads();
  float r = (threadIdx.x < 8) ? tmp[threadIdx.x] : 0.f;
  r += __shfl_xor(r, 4); r += __shfl_xor(r, 2); r += __shfl_xor(r, 1);
  return r;
}
__device__ __forceinline__ float blockReduceMax256(float v) {
  __shared__ float tmp[8];
  __syncthreads();
  v = fmaxf(v, __shfl_xor(v, 16)); v = fmaxf(v, __shfl_xor(v, 8));
  v = fmaxf(v, __shfl_xor(v, 4));  v = fmaxf(v, __shfl_xor(v, 2));
  v = fmaxf(v, __shfl_xor(v, 1));
  if ((threadIdx.x & 31) == 0) tmp[threadIdx.x >> 5] = v;
  __syncthreads();
  float r = (threadIdx.x < 8) ? tmp[threadIdx.x] : -1e30f;
  r = fmaxf(r, __shfl_xor(r, 4)); r = fmaxf(r, __shfl_xor(r, 2));
  r = fmaxf(r, __shfl_xor(r, 1));
  return r;
}

// ---------------------------------------------------------------------------
// Utility kernels
// ---------------------------------------------------------------------------
__global__ void k_zero_buf(float* p, long n) {
  long i = (long)blockIdx.x * blockDim.x + threadIdx.x;
  if (i < n) p[i] = 0.f;
}

__global__ void k_bias_softplus(float* X, const float* bias, int M, int N) {
  long i = (long)blockIdx.x * blockDim.x + threadIdx.x;
  if (i >= (long)M * N) return;
  int c = (int)(i % N);
  X[i] = softplusf(X[i] + bias[c]);
}

__global__ __launch_bounds__(256) void k_colstats(const float* X, int M, int N,
                                                  float* mean, float* var) {
  int j = blockIdx.x;
  float s1 = 0.f, s2 = 0.f;
  for (int i = threadIdx.x; i < M; i += 256) {
    float x = X[(size_t)i * N + j];
    s1 += x; s2 += x * x;
  }
  s1 = blockReduceSum256(s1);
  s2 = blockReduceSum256(s2);
  if (threadIdx.x == 0) {
    float m = s1 / M;
    mean[j] = m;
    var[j]  = s2 / M - m * m;
  }
}

__global__ void k_rownorm2(const float* X, int N, float* out) {
  int r = blockIdx.x;
  float s = 0.f;
  for (int c = threadIdx.x; c < N; c += 64) {
    float x = X[(size_t)r * N + c];
    s += x * x;
  }
  s += __shfl_xor(s, 16); s += __shfl_xor(s, 8); s += __shfl_xor(s, 4);
  s += __shfl_xor(s, 2);  s += __shfl_xor(s, 1);
  __shared__ float t[2];
  if ((threadIdx.x & 31) == 0) t[threadIdx.x >> 5] = s;
  __syncthreads();
  if (threadIdx.x == 0) out[r] = t[0] + t[1];
}

// ---------------------------------------------------------------------------
// Generic bf16 WMMA GEMM: C[M,N] = act(A[M,K] @ W + bias)
//   TRANSB=0: W is [K,N];  TRANSB=1: W is [N,K] (B = W^T)
//   grid.z > 1 => split-K, atomic f32 accumulate (bias/act skipped)
//   Block tile 128x64, 8 waves, K-step 32, fragment-major LDS staging.
// ---------------------------------------------------------------------------
template <int ACT, int TRANSB>
__global__ __launch_bounds__(256) void k_gemm(const float* __restrict__ A,
                                              const float* __restrict__ W,
                                              const float* __restrict__ bias,
                                              float* __restrict__ C,
                                              int M, int N, int K, int kchunk) {
  __shared__ __align__(32) __bf16 AshF[8][32][16];   // 8 m-tiles, frag-major
  __shared__ __align__(32) __bf16 BshF[4][32][16];   // 4 n-tiles, frag-major
  const int mbase = blockIdx.x * 128;
  const int nbase = blockIdx.y * 64;
  const int kstart = blockIdx.z * kchunk;
  const int kend = min(K, kstart + kchunk);
  const int tid = threadIdx.x, lane = tid & 31, wv = tid >> 5;
  const int h = lane >> 4, ml = lane & 15;

  v8f acc[4] = {};
  for (int k0 = kstart; k0 < kend; k0 += 32) {
    if (k0 + 32 < kend) {
      int pr = mbase + (tid >> 1);
      if (pr < M) __builtin_prefetch(&A[(size_t)pr * K + k0 + 32], 0, 1);
    }
    // stage A: 128x32 -> pairs (2048), packed b32 stores into frag layout
    for (int i = tid; i < 2048; i += 256) {
      int m_local = i >> 4;
      int kl = (i & 15) * 2;
      int gm = mbase + m_local, gk = k0 + kl;
      float a0 = (gm < M && gk < kend) ? A[(size_t)gm * K + gk] : 0.f;
      float a1 = (gm < M && gk + 1 < kend) ? A[(size_t)gm * K + gk + 1] : 0.f;
      int fh, fe; fragpos(kl, fh, fe);
      *(unsigned*)&AshF[m_local >> 4][fh * 16 + (m_local & 15)][fe] =
          pack_bf16(a0, a1);
    }
    // stage B: 32x64 -> pairs (1024)
    for (int i = tid; i < 1024; i += 256) {
      int n_local = i & 63;
      int kl = (i >> 6) * 2;
      int gk = k0 + kl, gn = nbase + n_local;
      float b0 = 0.f, b1 = 0.f;
      if (gn < N) {
        if (TRANSB) {
          if (gk < kend)     b0 = W[(size_t)gn * K + gk];
          if (gk + 1 < kend) b1 = W[(size_t)gn * K + gk + 1];
        } else {
          if (gk < kend)     b0 = W[(size_t)gk * N + gn];
          if (gk + 1 < kend) b1 = W[(size_t)(gk + 1) * N + gn];
        }
      }
      int fh, fe; fragpos(kl, fh, fe);
      *(unsigned*)&BshF[n_local >> 4][fh * 16 + (n_local & 15)][fe] =
          pack_bf16(b0, b1);
    }
    __syncthreads();
    v16bf af = *(const v16bf*)AshF[wv][lane];
#pragma unroll
    for (int nt = 0; nt < 4; ++nt) {
      v16bf bfv = *(const v16bf*)BshF[nt][lane];
      acc[nt] = __builtin_amdgcn_wmma_f32_16x16x32_bf16(
          false, af, false, bfv, (short)0, acc[nt], false, false);
    }
    __syncthreads();
  }
  const bool split = (gridDim.z > 1);
#pragma unroll
  for (int nt = 0; nt < 4; ++nt)
#pragma unroll
    for (int r = 0; r < 8; ++r) {
      int row = mbase + wv * 16 + r + 8 * h;
      int col = nbase + nt * 16 + ml;
      if (row < M && col < N) {
        float v = acc[nt][r];
        if (split) {
          atomicAdd(&C[(size_t)row * N + col], v);
        } else {
          if (bias) v += bias[col];
          if (ACT == 1) v = softplusf(v);
          C[(size_t)row * N + col] = v;
        }
      }
    }
}

// ---------------------------------------------------------------------------
// BN(train) + reparameterize + softmax(theta) + KL accumulation
// ---------------------------------------------------------------------------
__global__ void k_bn_reparam(const float* mu_raw, const float* lv_raw,
                             const float* mum, const float* muv,
                             const float* lvm, const float* lvv,
                             const float* mub, const float* lvb,
                             const float* epsn, float* zbuf, float* theta,
                             float* out) {
  int i = blockIdx.x * blockDim.x + threadIdx.x;
  if (i >= B_) return;
  const float var2 = 1.f - 1.f / (float)K_;
  const float logv2 = logf(var2);
  float kl = 0.f;
  for (int k = 0; k < K_; ++k) {
    size_t idx = (size_t)i * K_ + k;
    float mu = (mu_raw[idx] - mum[k]) * rsqrtf(muv[k] + BN_EPS) + mub[k];
    float lv = (lv_raw[idx] - lvm[k]) * rsqrtf(lvv[k] + BN_EPS) + lvb[k];
    float z = mu + epsn[idx] * __expf(0.5f * lv);
    zbuf[idx] = z;
    kl += __expf(lv) / var2 + mu * mu / var2 + (logv2 - lv);
  }
  atomicAdd(&out[1], 0.5f * (kl - (float)K_) * (1.f / (float)B_));
  float mx = -1e30f;
  for (int k = 0; k < K_; ++k) mx = fmaxf(mx, zbuf[(size_t)i * K_ + k]);
  float s = 0.f;
  for (int k = 0; k < K_; ++k) s += __expf(zbuf[(size_t)i * K_ + k] - mx);
  float inv = 1.f / s;
  for (int k = 0; k < K_; ++k)
    theta[(size_t)i * K_ + k] = __expf(zbuf[(size_t)i * K_ + k] - mx) * inv;
}

// ---------------------------------------------------------------------------
// dist_tw[100,V] = |t|^2 + |w|^2 - 2 t.w  (bf16 WMMA, frag-major staging)
// ---------------------------------------------------------------------------
__global__ __launch_bounds__(256) void k_dist(const float* __restrict__ topic,
                                              const float* __restrict__ word,
                                              const float* __restrict__ tn2,
                                              const float* __restrict__ wn2,
                                              float* __restrict__ dist) {
  __shared__ __align__(32) __bf16 AshF[7][32][16];  // 7 m-tiles (112 rows)
  __shared__ __align__(32) __bf16 BshF[8][32][16];  // 8 n-tiles (128 cols)
  const int colbase = blockIdx.x * 128;
  const int tid = threadIdx.x, lane = tid & 31, wv = tid >> 5;
  const int h = lane >> 4, ml = lane & 15;
  v8f acc[8] = {};
  for (int k0 = 0; k0 < E_; k0 += 32) {
    // A: topic[112,32] pairs (1792)
    for (int i = tid; i < 1792; i += 256) {
      int m_local = i >> 4;
      int kl = (i & 15) * 2;
      int gk = k0 + kl;
      float a0 = (m_local < K_ && gk < E_) ? topic[(size_t)m_local * E_ + gk] : 0.f;
      float a1 = (m_local < K_ && gk + 1 < E_) ? topic[(size_t)m_local * E_ + gk + 1] : 0.f;
      int fh, fe; fragpos(kl, fh, fe);
      *(unsigned*)&AshF[m_local >> 4][fh * 16 + (m_local & 15)][fe] =
          pack_bf16(a0, a1);
    }
    // B: word^T[32,128] pairs (2048); pair reads are contiguous in word row
    for (int i = tid; i < 2048; i += 256) {
      int n_local = i & 127;
      int kl = (i >> 7) * 2;
      int gk = k0 + kl, gc = colbase + n_local;
      float b0 = 0.f, b1 = 0.f;
      if (gc < V_) {
        if (gk < E_)     b0 = word[(size_t)gc * E_ + gk];
        if (gk + 1 < E_) b1 = word[(size_t)gc * E_ + gk + 1];
      }
      int fh, fe; fragpos(kl, fh, fe);
      *(unsigned*)&BshF[n_local >> 4][fh * 16 + (n_local & 15)][fe] =
          pack_bf16(b0, b1);
    }
    __syncthreads();
    if (wv < 7) {
      v16bf af = *(const v16bf*)AshF[wv][lane];
#pragma unroll
      for (int nt = 0; nt < 8; ++nt) {
        v16bf bfv = *(const v16bf*)BshF[nt][lane];
        acc[nt] = __builtin_amdgcn_wmma_f32_16x16x32_bf16(
            false, af, false, bfv, (short)0, acc[nt], false, false);
      }
    }
    __syncthreads();
  }
  if (wv < 7) {
#pragma unroll
    for (int nt = 0; nt < 8; ++nt)
#pragma unroll
      for (int r = 0; r < 8; ++r) {
        int m = wv * 16 + r + 8 * h;
        int col = colbase + nt * 16 + ml;
        if (m < K_ && col < V_)
          dist[(size_t)m * V_ + col] = tn2[m] + wn2[col] - 2.f * acc[nt][r];
      }
  }
}

// beta = softmax(-dist/0.2, axis=K)  and  Km = exp(-ALPHA*dist)
__global__ void k_beta_km(const float* dist, float* beta, float* Km) {
  int j = blockIdx.x * blockDim.x + threadIdx.x;
  if (j >= V_) return;
  float mx = -1e30f;
  for (int k = 0; k < K_; ++k) mx = fmaxf(mx, -dist[(size_t)k * V_ + j] * 5.f);
  float s = 0.f;
  for (int k = 0; k < K_; ++k) s += __expf(-dist[(size_t)k * V_ + j] * 5.f - mx);
  float inv = 1.f / s;
  for (int k = 0; k < K_; ++k) {
    float d = dist[(size_t)k * V_ + j];
    beta[(size_t)k * V_ + j] = __expf(-d * 5.f - mx) * inv;
    Km[(size_t)k * V_ + j] = __expf(-d * ALPHA_);
  }
}

// S = theta^T theta (100x100 Gram; BN variance trick)
__global__ void k_AtA(const float* theta, float* S) {
  int p = blockIdx.x * blockDim.x + threadIdx.x;
  if (p >= K_ * K_) return;
  int k = p / K_, l = p % K_;
  float s = 0.f;
  for (int i = 0; i < B_; ++i)
    s += theta[(size_t)i * K_ + k] * theta[(size_t)i * K_ + l];
  S[p] = s;
}

// per-column BN stats of Y = theta@beta without forming Y
__global__ __launch_bounds__(256) void k_beta_colstats(const float* beta,
                                                       const float* S,
                                                       const float* thmean,
                                                       float* colm, float* cols) {
  __shared__ float Ssh[K_ * K_];
  __shared__ float mth[K_];
  for (int i = threadIdx.x; i < K_ * K_; i += 256) Ssh[i] = S[i];
  for (int i = threadIdx.x; i < K_; i += 256) mth[i] = thmean[i];
  __syncthreads();
  int j = blockIdx.x * 256 + threadIdx.x;
  if (j >= V_) return;
  float bj[K_];
  for (int k = 0; k < K_; ++k) bj[k] = beta[(size_t)k * V_ + j];
  float m = 0.f;
  for (int k = 0; k < K_; ++k) m += mth[k] * bj[k];
  float q = 0.f;
  for (int k = 0; k < K_; ++k) {
    float t = 0.f;
    for (int l = 0; l < K_; ++l) t += Ssh[k * K_ + l] * bj[l];
    q += bj[k] * t;
  }
  q *= (1.f / (float)B_);
  float var = q - m * m;
  colm[j] = m;
  cols[j] = rsqrtf(var + BN_EPS);
}

// ---------------------------------------------------------------------------
// Fused recon: Y = theta@beta (WMMA), BN-normalize, online row-LSE and
// sum(data*x) without materializing [B,V]. Frag-major staging; theta block
// staged once for all 4 K-steps.
// ---------------------------------------------------------------------------
#define RC_CH     96
#define RC_NCHUNK 9
#define RC_SLICES 58
__global__ __launch_bounds__(256) void k_recon_main(
    const float* __restrict__ theta, const float* __restrict__ beta,
    const float* __restrict__ data, const float* __restrict__ colm,
    const float* __restrict__ cols, const float* __restrict__ dbias,
    float* __restrict__ partial) {
  __shared__ __align__(32) __bf16 AshF[8][4][32][16];  // 32 KB, staged once
  __shared__ __align__(32) __bf16 BshF[6][4][32][16];  // 24 KB per chunk
  __shared__ float cmS[RC_CH], csS[RC_CH], cbS[RC_CH];
  const int rb = blockIdx.x, slice = blockIdx.y;
  const int tid = threadIdx.x, lane = tid & 31, wv = tid >> 5;
  const int h = lane >> 4, ml = lane & 15;

  // stage theta block [128 x 128(Kpad)] in fragment-major form
  for (int i = tid; i < 8192; i += 256) {
    int m_local = i >> 6;
    int k = (i & 63) * 2;
    float a0 = (k < K_) ? theta[(size_t)(rb * 128 + m_local) * K_ + k] : 0.f;
    float a1 = (k + 1 < K_) ? theta[(size_t)(rb * 128 + m_local) * K_ + k + 1] : 0.f;
    int fh, fe; fragpos(k & 31, fh, fe);
    *(unsigned*)&AshF[m_local >> 4][k >> 5][fh * 16 + (m_local & 15)][fe] =
        pack_bf16(a0, a1);
  }
  __syncthreads();

  float gmax[8], gsum[8], gdot[8], gds[8];
#pragma unroll
  for (int r = 0; r < 8; ++r) { gmax[r] = -1e30f; gsum[r] = 0.f; gdot[r] = 0.f; gds[r] = 0.f; }

  for (int ch = 0; ch < RC_NCHUNK; ++ch) {
    int cb = (slice * RC_NCHUNK + ch) * RC_CH;
    if (cb >= V_) break;
    // stage beta chunk [128(Kpad) x 96] frag-major: 6144 pairs
    for (int i = tid; i < 6144; i += 256) {
      int n_local = i % RC_CH;
      int k = (i / RC_CH) * 2;
      int col = cb + n_local;
      float b0 = 0.f, b1 = 0.f;
      if (col < V_) {
        if (k < K_)     b0 = beta[(size_t)k * V_ + col];
        if (k + 1 < K_) b1 = beta[(size_t)(k + 1) * V_ + col];
      }
      int fh, fe; fragpos(k & 31, fh, fe);
      *(unsigned*)&BshF[n_local >> 4][k >> 5][fh * 16 + (n_local & 15)][fe] =
          pack_bf16(b0, b1);
    }
    if (tid < RC_CH) {
      int col = cb + tid;
      bool ok = col < V_;
      cmS[tid] = ok ? colm[col] : 0.f;
      csS[tid] = ok ? cols[col] : 0.f;
      cbS[tid] = ok ? dbias[col] : 0.f;
    }
    __syncthreads();

    v8f acc[6] = {};
#pragma unroll
    for (int ks = 0; ks < 4; ++ks) {
      v16bf af = *(const v16bf*)AshF[wv][ks][lane];
#pragma unroll
      for (int nt = 0; nt < 6; ++nt) {
        v16bf bfv = *(const v16bf*)BshF[nt][ks][lane];
        acc[nt] = __builtin_amdgcn_wmma_f32_16x16x32_bf16(
            false, af, false, bfv, (short)0, acc[nt], false, false);
      }
    }
    // epilogue: x = (Y - m)*s + b, fused data.x + online LSE
    float xs[6][8];
#pragma unroll
    for (int nt = 0; nt < 6; ++nt)
#pragma unroll
      for (int r = 0; r < 8; ++r) {
        int lc = nt * 16 + ml;
        int col = cb + lc;
        int row = rb * 128 + wv * 16 + r + 8 * h;
        float x = (acc[nt][r] - cmS[lc]) * csS[lc] + cbS[lc];
        if (col < V_) {
          float d = data[(size_t)row * V_ + col];
          gdot[r] += d * x;
          gds[r] += d;
          xs[nt][r] = x;
        } else {
          xs[nt][r] = -1e30f;
        }
      }
#pragma unroll
    for (int r = 0; r < 8; ++r) {
      float cm = xs[0][r];
#pragma unroll
      for (int nt = 1; nt < 6; ++nt) cm = fmaxf(cm, xs[nt][r]);
      cm = fmaxf(cm, __shfl_xor(cm, 1, 16));
      cm = fmaxf(cm, __shfl_xor(cm, 2, 16));
      cm = fmaxf(cm, __shfl_xor(cm, 4, 16));
      cm = fmaxf(cm, __shfl_xor(cm, 8, 16));
      float es = 0.f;
#pragma unroll
      for (int nt = 0; nt < 6; ++nt) es += __expf(xs[nt][r] - cm);
      es += __shfl_xor(es, 1, 16);
      es += __shfl_xor(es, 2, 16);
      es += __shfl_xor(es, 4, 16);
      es += __shfl_xor(es, 8, 16);
      float nm = fmaxf(gmax[r], cm);
      gsum[r] = gsum[r] * __expf(gmax[r] - nm) + es * __expf(cm - nm);
      gmax[r] = nm;
    }
    __syncthreads();
  }
#pragma unroll
  for (int r = 0; r < 8; ++r) {
    float d = gdot[r], s = gds[r];
    d += __shfl_xor(d, 1, 16); d += __shfl_xor(d, 2, 16);
    d += __shfl_xor(d, 4, 16); d += __shfl_xor(d, 8, 16);
    s += __shfl_xor(s, 1, 16); s += __shfl_xor(s, 2, 16);
    s += __shfl_xor(s, 4, 16); s += __shfl_xor(s, 8, 16);
    if (ml == 0) {
      int row = rb * 128 + wv * 16 + r + 8 * h;
      size_t base = ((size_t)row * RC_SLICES + slice) * 4;
      partial[base + 0] = gmax[r];
      partial[base + 1] = gsum[r];
      partial[base + 2] = d;
      partial[base + 3] = s;
    }
  }
}

__global__ void k_recon_combine(const float* partial, float* out) {
  int row = blockIdx.x * blockDim.x + threadIdx.x;
  if (row >= B_) return;
  float gm = -1e30f, gs = 0.f, dot = 0.f, ds = 0.f;
  for (int s = 0; s < RC_SLICES; ++s) {
    size_t base = ((size_t)row * RC_SLICES + s) * 4;
    float m = partial[base], sm = partial[base + 1];
    float nm = fmaxf(gm, m);
    gs = gs * __expf(gm - nm) + sm * __expf(m - nm);
    gm = nm;
    dot += partial[base + 2];
    ds += partial[base + 3];
  }
  float lse = gm + logf(gs);
  atomicAdd(&out[1], -(dot - ds * lse) * (1.f / (float)B_));
}

// ---------------------------------------------------------------------------
// ECR Sinkhorn on Km[100,V] (L2-resident: 20 MB << 192 MB)
// ---------------------------------------------------------------------------
__global__ void k_ecr_init(float* u) {
  if (threadIdx.x < K_) u[threadIdx.x] = 1.f / (float)K_;
}
__global__ void k_ecr_v(const float* Km, const float* u, float* v) {
  __shared__ float ush[K_];
  for (int i = threadIdx.x; i < K_; i += 256) ush[i] = u[i];
  __syncthreads();
  int j = blockIdx.x * 256 + threadIdx.x;
  if (j >= V_) return;
  float s = 0.f;
  for (int k = 0; k < K_; ++k) s += Km[(size_t)k * V_ + j] * ush[k];
  v[j] = (1.f / (float)V_) / (s + SK_EPS);
}
__global__ __launch_bounds__(256) void k_ecr_u(const float* Km, const float* v,
                                               float* u) {
  int k = blockIdx.x;
  float s = 0.f;
  for (int j = threadIdx.x; j < V_; j += 256) s += Km[(size_t)k * V_ + j] * v[j];
  s = blockReduceSum256(s);
  if (threadIdx.x == 0) u[k] = (1.f / (float)K_) / (s + SK_EPS);
}
__global__ __launch_bounds__(256) void k_ecr_loss(const float* Km,
                                                  const float* dist,
                                                  const float* u,
                                                  const float* v, float* out) {
  int k = blockIdx.y;
  int j = blockIdx.x * 256 + threadIdx.x;
  float s = 0.f;
  if (j < V_) {
    size_t idx = (size_t)k * V_ + j;
    s = u[k] * Km[idx] * v[j] * dist[idx];
  }
  s = blockReduceSum256(s);
  if (threadIdx.x == 0) atomicAdd(&out[2], 250.f * s);
}

// ---------------------------------------------------------------------------
// Small Sinkhorn (DCR / TCR): one workgroup, 20 clusters, 100 iterations.
// ---------------------------------------------------------------------------
__global__ __launch_bounds__(256) void k_sinkhorn_small(
    const float* __restrict__ X, int R, const float* __restrict__ Cl,
    const float* __restrict__ a, const float* __restrict__ bmass,
    float* Mws, float* Kws, float* uws, float weight, float* out, int outidx,
    const int* gate) {
  if (gate && *gate <= 10) return;
  __shared__ float cn2[G_], vsh[G_];
  __shared__ float vpart[256][G_];
  if (threadIdx.x < G_) {
    float s = 0.f;
    for (int p = 0; p < P_; ++p) {
      float c = Cl[(size_t)threadIdx.x * P_ + p];
      s += c * c;
    }
    cn2[threadIdx.x] = s;
  }
  __syncthreads();
  for (int r = threadIdx.x; r < R; r += 256) {
    float xn = 0.f;
    for (int p = 0; p < P_; ++p) {
      float x = X[(size_t)r * P_ + p];
      xn += x * x;
    }
    for (int c = 0; c < G_; ++c) {
      float d = 0.f;
      for (int p = 0; p < P_; ++p)
        d += X[(size_t)r * P_ + p] * Cl[(size_t)c * P_ + p];
      float m = xn + cn2[c] - 2.f * d;
      Mws[r * G_ + c] = m;
      Kws[r * G_ + c] = __expf(-ALPHA_ * m);
    }
    uws[r] = 1.f / (float)R;
  }
  __syncthreads();
  for (int it = 0; it < 100; ++it) {
    float p[G_];
#pragma unroll
    for (int c = 0; c < G_; ++c) p[c] = 0.f;
    for (int r = threadIdx.x; r < R; r += 256) {
      float ur = uws[r];
#pragma unroll
      for (int c = 0; c < G_; ++c) p[c] += Kws[r * G_ + c] * ur;
    }
#pragma unroll
    for (int c = 0; c < G_; ++c) vpart[threadIdx.x][c] = p[c];
    __syncthreads();
    for (int off = 128; off >= 1; off >>= 1) {
      if (threadIdx.x < off)
        for (int c = 0; c < G_; ++c)
          vpart[threadIdx.x][c] += vpart[threadIdx.x + off][c];
      __syncthreads();
    }
    if (threadIdx.x < G_)
      vsh[threadIdx.x] = bmass[threadIdx.x] / (vpart[0][threadIdx.x] + SK_EPS);
    __syncthreads();
    for (int r = threadIdx.x; r < R; r += 256) {
      float s = 0.f;
#pragma unroll
      for (int c = 0; c < G_; ++c) s += Kws[r * G_ + c] * vsh[c];
      uws[r] = a[r] / (s + SK_EPS);
    }
    __syncthreads();
  }
  float s = 0.f;
  for (int r = threadIdx.x; r < R; r += 256) {
    float ur = uws[r];
#pragma unroll
    for (int c = 0; c < G_; ++c)
      s += ur * Kws[r * G_ + c] * vsh[c] * Mws[r * G_ + c];
  }
  s = blockReduceSum256(s);
  if (threadIdx.x == 0) atomicAdd(&out[outidx], weight * s);
}

// ---------------------------------------------------------------------------
// NCE: per-row softmax of cosine matrix (dot computed via WMMA)
// ---------------------------------------------------------------------------
__global__ __launch_bounds__(256) void k_nce(const float* cosbuf,
                                             const float* pn2,
                                             const float* bn2, float* out) {
  int i = blockIdx.x;
  float rp = rsqrtf(pn2[i]);
  float mx = -1e30f;
  for (int j = threadIdx.x; j < B_; j += 256) {
    float c = cosbuf[(size_t)i * B_ + j] * rp * rsqrtf(bn2[j]);
    mx = fmaxf(mx, c);
  }
  mx = blockReduceMax256(mx);
  __shared__ float mxs, diag;
  if (threadIdx.x == 0) mxs = mx;
  __syncthreads();
  float se = 0.f;
  for (int j = threadIdx.x; j < B_; j += 256) {
    float c = cosbuf[(size_t)i * B_ + j] * rp * rsqrtf(bn2[j]);
    se += __expf(c - mxs);
    if (j == i) diag = c;
  }
  se = blockReduceSum256(se);
  if (threadIdx.x == 0) {
    float l = -(diag - mxs - logf(se));
    atomicAdd(&out[5], 10.f * l * (1.f / (float)B_));
  }
}

__global__ void k_finalize(float* out) {
  out[0] = out[1] + out[2] + out[3] + out[4] + out[5];
}

// ---------------------------------------------------------------------------
// Host launcher
// ---------------------------------------------------------------------------
extern "C" void kernel_launch(void* const* d_in, const int* in_sizes, int n_in,
                              void* d_out, int out_size, void* d_ws,
                              size_t ws_size, hipStream_t stream) {
  (void)in_sizes; (void)n_in; (void)out_size; (void)ws_size;
  const float* data    = (const float*)d_in[0];
  const float* bert    = (const float*)d_in[1];
  const float* epsn    = (const float*)d_in[2];
  const float* fc11w   = (const float*)d_in[3];
  const float* fc11b   = (const float*)d_in[4];
  const float* fc12w   = (const float*)d_in[5];
  const float* fc12b   = (const float*)d_in[6];
  const float* fc21w   = (const float*)d_in[7];
  const float* fc21b   = (const float*)d_in[8];
  const float* fc22w   = (const float*)d_in[9];
  const float* fc22b   = (const float*)d_in[10];
  const float* mubnb   = (const float*)d_in[11];
  const float* lvbnb   = (const float*)d_in[12];
  const float* decbnb  = (const float*)d_in[13];
  const float* wordemb = (const float*)d_in[14];
  const float* topicemb= (const float*)d_in[15];
  const float* thprjw  = (const float*)d_in[16];
  const float* thprjb  = (const float*)d_in[17];
  const float* tpprjw  = (const float*)d_in[18];
  const float* tpprjb  = (const float*)d_in[19];
  const float* prw     = (const float*)d_in[20];
  const float* prb     = (const float*)d_in[21];
  const float* clemb   = (const float*)d_in[22];
  const float* clmass  = (const float*)d_in[23];
  const float* docw    = (const float*)d_in[24];
  const float* topw    = (const float*)d_in[25];
  const int*   epoch   = (const int*)d_in[26];
  float* out = (float*)d_out;

  float* p = (float*)d_ws;
  auto alloc = [&](size_t n) { float* r = p; p += n; return r; };
  float* e1      = alloc((size_t)B_ * EU_);
  float* e2      = alloc((size_t)B_ * EU_);
  float* mu_raw  = alloc((size_t)B_ * K_);
  float* lv_raw  = alloc((size_t)B_ * K_);
  float* zbuf    = alloc((size_t)B_ * K_);
  float* theta   = alloc((size_t)B_ * K_);
  float* mum = alloc(K_); float* muv = alloc(K_);
  float* lvm = alloc(K_); float* lvv = alloc(K_);
  float* thm = alloc(K_); float* thv = alloc(K_);
  float* tn2     = alloc(K_);
  float* S       = alloc((size_t)K_ * K_);
  float* wn2     = alloc(V_);
  float* colm    = alloc(V_);
  float* colsv   = alloc(V_);
  float* dist    = alloc((size_t)K_ * V_);
  float* Km      = alloc((size_t)K_ * V_);
  float* beta    = alloc((size_t)K_ * V_);
  float* u_ecr   = alloc(K_);
  float* v_ecr   = alloc(V_);
  float* theta_p = alloc((size_t)B_ * P_);
  float* topic_p = alloc((size_t)K_ * P_);
  float* pbow    = alloc((size_t)B_ * P_);
  float* dcrM    = alloc((size_t)B_ * G_);
  float* dcrK    = alloc((size_t)B_ * G_);
  float* dcru    = alloc(B_);
  float* tcrM    = alloc((size_t)K_ * G_);
  float* tcrK    = alloc((size_t)K_ * G_);
  float* tcru    = alloc(K_);
  float* pn2     = alloc(B_);
  float* bn2     = alloc(B_);
  float* cosbuf  = alloc((size_t)B_ * B_);
  float* partial = alloc((size_t)B_ * RC_SLICES * 4);

  k_zero_buf<<<1, 64, 0, stream>>>(out, 6);
  k_zero_buf<<<(B_ * EU_ + 255) / 256, 256, 0, stream>>>(e1, (long)B_ * EU_);

  // --- encoder ---
  k_gemm<0, 0><<<dim3(8, 4, 16), 256, 0, stream>>>(data, fc11w, nullptr, e1,
                                                   B_, EU_, V_, 3125);
  k_bias_softplus<<<(B_ * EU_ + 255) / 256, 256, 0, stream>>>(e1, fc11b, B_, EU_);
  k_gemm<1, 0><<<dim3(8, 4, 1), 256, 0, stream>>>(e1, fc12w, fc12b, e2,
                                                  B_, EU_, EU_, EU_);
  k_gemm<0, 0><<<dim3(8, 2, 1), 256, 0, stream>>>(e2, fc21w, fc21b, mu_raw,
                                                  B_, K_, EU_, EU_);
  k_gemm<0, 0><<<dim3(8, 2, 1), 256, 0, stream>>>(e2, fc22w, fc22b, lv_raw,
                                                  B_, K_, EU_, EU_);
  k_colstats<<<K_, 256, 0, stream>>>(mu_raw, B_, K_, mum, muv);
  k_colstats<<<K_, 256, 0, stream>>>(lv_raw, B_, K_, lvm, lvv);
  k_bn_reparam<<<4, 256, 0, stream>>>(mu_raw, lv_raw, mum, muv, lvm, lvv,
                                      mubnb, lvbnb, epsn, zbuf, theta, out);
  k_colstats<<<K_, 256, 0, stream>>>(theta, B_, K_, thm, thv);

  // --- topic/word distance, beta, Km ---
  k_rownorm2<<<K_, 64, 0, stream>>>(topicemb, E_, tn2);
  k_rownorm2<<<V_, 64, 0, stream>>>(wordemb, E_, wn2);
  k_dist<<<(V_ + 127) / 128, 256, 0, stream>>>(topicemb, wordemb, tn2, wn2, dist);
  k_beta_km<<<(V_ + 255) / 256, 256, 0, stream>>>(dist, beta, Km);

  // --- recon ---
  k_AtA<<<(K_ * K_ + 255) / 256, 256, 0, stream>>>(theta, S);
  k_beta_colstats<<<(V_ + 255) / 256, 256, 0, stream>>>(beta, S, thm, colm, colsv);
  k_recon_main<<<dim3(8, RC_SLICES), 256, 0, stream>>>(theta, beta, data, colm,
                                                       colsv, decbnb, partial);
  k_recon_combine<<<4, 256, 0, stream>>>(partial, out);

  // --- ECR sinkhorn ---
  k_ecr_init<<<1, 128, 0, stream>>>(u_ecr);
  for (int it = 0; it < 100; ++it) {
    k_ecr_v<<<(V_ + 255) / 256, 256, 0, stream>>>(Km, u_ecr, v_ecr);
    k_ecr_u<<<K_, 256, 0, stream>>>(Km, v_ecr, u_ecr);
  }
  k_ecr_loss<<<dim3((V_ + 255) / 256, K_), 256, 0, stream>>>(Km, dist, u_ecr,
                                                             v_ecr, out);

  // --- DCR ---
  k_gemm<0, 0><<<dim3(8, 6, 1), 256, 0, stream>>>(theta, thprjw, thprjb,
                                                  theta_p, B_, P_, K_, K_);
  k_sinkhorn_small<<<1, 256, 0, stream>>>(theta_p, B_, clemb, docw, clmass,
                                          dcrM, dcrK, dcru, 250.f, out, 3,
                                          nullptr);
  // --- TCR (epoch-gated on device) ---
  k_gemm<0, 0><<<dim3(1, 6, 1), 256, 0, stream>>>(topicemb, tpprjw, tpprjb,
                                                  topic_p, K_, P_, E_, E_);
  k_sinkhorn_small<<<1, 256, 0, stream>>>(topic_p, K_, clemb, topw, clmass,
                                          tcrM, tcrK, tcru, 250.f, out, 4,
                                          epoch);

  // --- InfoNCE ---
  k_gemm<0, 0><<<dim3(8, 6, 1), 256, 0, stream>>>(theta, prw, prb, pbow,
                                                  B_, P_, K_, K_);
  k_rownorm2<<<B_, 64, 0, stream>>>(pbow, P_, pn2);
  k_rownorm2<<<B_, 64, 0, stream>>>(bert, P_, bn2);
  k_gemm<0, 1><<<dim3(8, 16, 1), 256, 0, stream>>>(pbow, bert, nullptr, cosbuf,
                                                   B_, B_, P_, P_);
  k_nce<<<B_, 256, 0, stream>>>(cosbuf, pn2, bn2, out);

  k_finalize<<<1, 1, 0, stream>>>(out);
}
